// MultiHeadAttention_26001732010562
// MI455X (gfx1250) — matrix operand order repaired and checked by reading, not compiled, not run on hardware
//
#include <hip/hip_runtime.h>

// ---------------------------------------------------------------------------
// MI455X (gfx1250) multi-head attention, bf16 WMMA everywhere.
// ---------------------------------------------------------------------------

typedef __attribute__((ext_vector_type(16))) __bf16 v16bf;
typedef __attribute__((ext_vector_type(8)))  __bf16 bf16x8;
typedef __attribute__((ext_vector_type(8)))  float  v8f;

#define BATCH   4
#define SEQ     2048
#define DMODEL  1024
#define NH      16
#define DK      64
#define MTOT    (BATCH * SEQ)          // 8192 rows
#define KDIM    DMODEL                 // 1024 contraction for projections

// 1/sqrt(dk) * log2(e): folded into Q projection so scores are in exp2 domain
#define ALPHA_Q (0.125f * 1.4426950408889634f)

static __device__ __forceinline__ bf16x8 ld8(const __bf16* p) {
    return *reinterpret_cast<const bf16x8*>(p);
}

static __device__ __forceinline__ v16bf mk16(bf16x8 lo, bf16x8 hi) {
    v16bf r;
#pragma unroll
    for (int i = 0; i < 8; ++i) { r[i] = lo[i]; r[i + 8] = hi[i]; }
    return r;
}

static __device__ __forceinline__ v8f wmma_bf16(v16bf a, v16bf b, v8f c) {
    // D = A(16x32) * B(32x16) + C, f32 accumulate
    return __builtin_amdgcn_wmma_f32_16x16x32_bf16(
        /*neg_a=*/false, a, /*neg_b=*/false, b,
        /*c_mod=*/(short)0, c, /*reuse_a=*/false, /*reuse_b=*/false);
}

// xor-lane-swap within group of 32 via ds_swizzle (masks <=8 stay in 16-lane half)
template <int MASK>
static __device__ __forceinline__ float swz_xor(float x) {
    return __int_as_float(
        __builtin_amdgcn_ds_swizzle(__float_as_int(x), (MASK << 10) | 0x1f));
}

// ---------------------------------------------------------------------------
// Elementwise f32 -> bf16 cast
// ---------------------------------------------------------------------------
__global__ __launch_bounds__(256) void cast_bf16_kernel(
    const float* __restrict__ in, __bf16* __restrict__ out, int n) {
    int i = blockIdx.x * 256 + threadIdx.x;
    if (i < n) out[i] = (__bf16)in[i];
}

// ---------------------------------------------------------------------------
// Transpose-cast: W[k][n] (f32, KDIM x DMODEL) -> Wt[n][k] (bf16)
// ---------------------------------------------------------------------------
__global__ __launch_bounds__(256) void transpose_cast_kernel(
    const float* __restrict__ W, __bf16* __restrict__ Wt) {
    int i = blockIdx.x * 256 + threadIdx.x;   // over DMODEL*KDIM
    int n = i >> 10;                          // /1024
    int k = i & 1023;
    Wt[i] = (__bf16)W[k * DMODEL + n];
}

// ---------------------------------------------------------------------------
// GEMM: C = (A[M,K](bf16) * Bt[N,K](bf16, pre-transposed) + bias) * alpha
// Each wave computes a 32x64 tile (2 row-subtiles x 4 col-subtiles),
// 8 v_wmma_f32_16x16x32_bf16 per K-step of 32.
// MODE 0: store bf16 [M,N]        (Q/K projections; alpha folds softmax scale)
// MODE 1: store bf16 Vt[b,h,d,s]  (V projection, transposed per head)
// MODE 2: store f32  [M,N] to d_out (output projection)
// ---------------------------------------------------------------------------
template <int MODE>
__global__ __launch_bounds__(256) void gemm_wmma_kernel(
    const __bf16* __restrict__ A, const __bf16* __restrict__ Bt,
    const float* __restrict__ bias, void* __restrict__ Cout, float alpha) {

    const int lane = threadIdx.x & 31;
    const int wave = threadIdx.x >> 5;
    const int gw   = blockIdx.x * 8 + wave;       // global wave-tile id
    const int nCB  = DMODEL / 64;                 // 16 col-blocks
    const int rb   = gw / nCB;                    // 0..255 (MTOT/32)
    const int cb   = gw % nCB;
    const int m0   = rb * 32;
    const int n0   = cb * 64;
    const int lh   = lane >> 4;                   // lane half
    const int ln   = lane & 15;

    v8f acc[2][4];
#pragma unroll
    for (int mi = 0; mi < 2; ++mi)
#pragma unroll
        for (int j = 0; j < 4; ++j)
#pragma unroll
            for (int r = 0; r < 8; ++r) acc[mi][j][r] = 0.0f;

    const __bf16* arow0 = A + (size_t)(m0 + ln) * KDIM;
    const __bf16* arow1 = A + (size_t)(m0 + 16 + ln) * KDIM;
    const __bf16* brow[4];
#pragma unroll
    for (int j = 0; j < 4; ++j)
        brow[j] = Bt + (size_t)(n0 + j * 16 + ln) * KDIM;

    for (int k0 = 0; k0 < KDIM; k0 += 32) {
        // A fragment: lanes 0-15 rows m, K {0..7,16..23}; lanes 16-31 K {8..15,24..31}
        v16bf a0 = mk16(ld8(arow0 + k0 + lh * 8), ld8(arow0 + k0 + 16 + lh * 8));
        v16bf a1 = mk16(ld8(arow1 + k0 + lh * 8), ld8(arow1 + k0 + 16 + lh * 8));
#pragma unroll
        for (int j = 0; j < 4; ++j) {
            // B fragment: lanes 0-15 K {0..7,16..23}, lanes 16-31 K {8..15,24..31}
            v16bf b = mk16(ld8(brow[j] + k0 + lh * 8),
                           ld8(brow[j] + k0 + 16 + lh * 8));
            acc[0][j] = wmma_bf16(a0, b, acc[0][j]);
            acc[1][j] = wmma_bf16(a1, b, acc[1][j]);
        }
    }

#pragma unroll
    for (int mi = 0; mi < 2; ++mi) {
#pragma unroll
        for (int j = 0; j < 4; ++j) {
            const int ng = n0 + j * 16 + ln;
            const float bv = bias[ng];
#pragma unroll
            for (int r = 0; r < 8; ++r) {
                const int mg = m0 + mi * 16 + r + lh * 8;   // C layout: M = r + 8*half
                const float v = (acc[mi][j][r] + bv) * alpha;
                if (MODE == 0) {
                    ((__bf16*)Cout)[(size_t)mg * DMODEL + ng] = (__bf16)v;
                } else if (MODE == 1) {
                    const int b = mg / SEQ, s = mg % SEQ;
                    const int h = ng / DK,  d = ng % DK;
                    ((__bf16*)Cout)[(((size_t)b * NH + h) * DK + d) * SEQ + s] = (__bf16)v;
                } else {
                    ((float*)Cout)[(size_t)mg * DMODEL + ng] = v;
                }
            }
        }
    }
}

// ---------------------------------------------------------------------------
// Flash attention: one wave per (b, h, 16-row q block). Online softmax over
// 64-key blocks, scores already in exp2 domain (Q pre-scaled by alpha).
// Row sums computed with an extra WMMA against an all-ones B matrix (no
// cross-lane shuffles); row max reduced with 4 ds_swizzle xor steps.
// Qp, Kp: bf16 [B*S, D]; Vt: bf16 [B,H,DK,S]; ctx out: bf16 [B*S, D].
// ---------------------------------------------------------------------------
__global__ __launch_bounds__(256) void flash_attn_kernel(
    const __bf16* __restrict__ Qp, const __bf16* __restrict__ Kp,
    const __bf16* __restrict__ Vt, __bf16* __restrict__ ctx) {

    __shared__ __bf16 pbuf[8][16 * 64];            // per-wave P staging (2 KiB each)

    const int lane = threadIdx.x & 31;
    const int wave = threadIdx.x >> 5;
    const int gw   = blockIdx.x * 8 + wave;
    const int qpb  = SEQ / 16;                     // 128 q-blocks per (b,h)
    const int b    = gw / (NH * qpb);
    const int rem  = gw % (NH * qpb);
    const int h    = rem / qpb;
    const int q0   = (rem % qpb) * 16;
    const int lh   = lane >> 4;
    const int ln   = lane & 15;

    // Q A-fragments for the full dk=64 (two K-steps of 32), rows q0..q0+15
    const __bf16* qrow = Qp + (size_t)(b * SEQ + q0 + ln) * DMODEL + h * DK;
    const v16bf aq0 = mk16(ld8(qrow + lh * 8),      ld8(qrow + 16 + lh * 8));
    const v16bf aq1 = mk16(ld8(qrow + 32 + lh * 8), ld8(qrow + 48 + lh * 8));

    // all-ones B fragment for row-sum WMMA
    v16bf ones;
#pragma unroll
    for (int i = 0; i < 16; ++i) ones[i] = (__bf16)1.0f;

    float mrun[8], lrun[8];
    v8f acc[4];
#pragma unroll
    for (int r = 0; r < 8; ++r) { mrun[r] = -3.0e38f; lrun[r] = 0.0f; }
#pragma unroll
    for (int j = 0; j < 4; ++j)
#pragma unroll
        for (int r = 0; r < 8; ++r) acc[j][r] = 0.0f;

    __bf16* myp = &pbuf[wave][0];

    for (int kb = 0; kb < SEQ; kb += 64) {
        // ---- scores (log2 domain) for 64 keys: four 16x16 tiles ----
        v8f p[4];
#pragma unroll
        for (int t = 0; t < 4; ++t) {
            const __bf16* krow =
                Kp + (size_t)(b * SEQ + kb + t * 16 + ln) * DMODEL + h * DK;
            v16bf bk0 = mk16(ld8(krow + lh * 8),      ld8(krow + 16 + lh * 8));
            v16bf bk1 = mk16(ld8(krow + 32 + lh * 8), ld8(krow + 48 + lh * 8));
            v8f c = {0.f, 0.f, 0.f, 0.f, 0.f, 0.f, 0.f, 0.f};
            c = wmma_bf16(aq0, bk0, c);
            c = wmma_bf16(aq1, bk1, c);
            p[t] = c;
        }

        // ---- running max update (rows at vgpr r, N across lane half) ----
        float mnew[8], sc[8];
#pragma unroll
        for (int r = 0; r < 8; ++r) {
            float mx = fmaxf(fmaxf(p[0][r], p[1][r]), fmaxf(p[2][r], p[3][r]));
            mx = fmaxf(mx, swz_xor<1>(mx));
            mx = fmaxf(mx, swz_xor<2>(mx));
            mx = fmaxf(mx, swz_xor<4>(mx));
            mx = fmaxf(mx, swz_xor<8>(mx));
            mnew[r] = fmaxf(mrun[r], mx);
            sc[r]   = exp2f(mrun[r] - mnew[r]);
            mrun[r] = mnew[r];
        }

        // ---- P = exp2(s - mnew), staged to LDS (16x64 bf16 row-major) ----
#pragma unroll
        for (int t = 0; t < 4; ++t)
#pragma unroll
            for (int r = 0; r < 8; ++r) {
                const float e = exp2f(p[t][r] - mnew[r]);
                myp[(r + lh * 8) * 64 + t * 16 + ln] = (__bf16)e;
            }

        // ---- read P as two A fragments (k 0..31 and 32..63) ----
        const v16bf pa0 = mk16(ld8(myp + ln * 64 + lh * 8),
                               ld8(myp + ln * 64 + 16 + lh * 8));
        const v16bf pa1 = mk16(ld8(myp + ln * 64 + 32 + lh * 8),
                               ld8(myp + ln * 64 + 48 + lh * 8));

        // ---- row sums via WMMA against ones: every lane of row r gets sum ----
        v8f rs = {0.f, 0.f, 0.f, 0.f, 0.f, 0.f, 0.f, 0.f};
        rs = wmma_bf16(pa0, ones, rs);
        rs = wmma_bf16(pa1, ones, rs);
#pragma unroll
        for (int r = 0; r < 8; ++r) lrun[r] = lrun[r] * sc[r] + rs[r];

        // ---- rescale accumulators, then acc += P x V (4 d-tiles, 2 k-steps) ----
#pragma unroll
        for (int j = 0; j < 4; ++j)
#pragma unroll
            for (int r = 0; r < 8; ++r) acc[j][r] *= sc[r];

#pragma unroll
        for (int j = 0; j < 4; ++j) {
            const __bf16* vrow =
                Vt + ((size_t)(b * NH + h) * DK + j * 16 + ln) * SEQ + kb + lh * 8;
            v16bf vb0 = mk16(ld8(vrow),      ld8(vrow + 16));
            v16bf vb1 = mk16(ld8(vrow + 32), ld8(vrow + 48));
            acc[j] = wmma_bf16(pa0, vb0, acc[j]);
            acc[j] = wmma_bf16(pa1, vb1, acc[j]);
        }
    }

    // ---- normalize and write context (bf16 [B*S, D]) ----
    float inv[8];
#pragma unroll
    for (int r = 0; r < 8; ++r) inv[r] = 1.0f / lrun[r];
#pragma unroll
    for (int j = 0; j < 4; ++j)
#pragma unroll
        for (int r = 0; r < 8; ++r) {
            const size_t row = (size_t)(b * SEQ + q0 + r + lh * 8);
            ctx[row * DMODEL + h * DK + j * 16 + ln] = (__bf16)(acc[j][r] * inv[r]);
        }
}

// ---------------------------------------------------------------------------
// Host orchestration
// ---------------------------------------------------------------------------
extern "C" void kernel_launch(void* const* d_in, const int* in_sizes, int n_in,
                              void* d_out, int out_size, void* d_ws, size_t ws_size,
                              hipStream_t stream) {
    const float* query = (const float*)d_in[0];
    const float* key   = (const float*)d_in[1];
    const float* value = (const float*)d_in[2];
    const float* Wq    = (const float*)d_in[3];
    const float* bq    = (const float*)d_in[4];
    const float* Wk    = (const float*)d_in[5];
    const float* bk    = (const float*)d_in[6];
    const float* Wv    = (const float*)d_in[7];
    const float* bv    = (const float*)d_in[8];
    const float* Wo    = (const float*)d_in[9];
    const float* bo    = (const float*)d_in[10];

    const size_t MB = 1024ull * 1024ull;
    char* ws = (char*)d_ws;
    __bf16* Xb = (__bf16*)(ws + 0 * MB);    // 16 MiB activation staging (reused)
    __bf16* Wt = (__bf16*)(ws + 16 * MB);   //  2 MiB transposed weights (reused)
    __bf16* Qp = (__bf16*)(ws + 18 * MB);   // 16 MiB
    __bf16* Kp = (__bf16*)(ws + 34 * MB);   // 16 MiB
    __bf16* Vt = (__bf16*)(ws + 50 * MB);   // 16 MiB (per-head transposed)
    __bf16* Cx = (__bf16*)(ws + 66 * MB);   // 16 MiB attention context

    const int NEL = MTOT * DMODEL;                    // 8,388,608
    const int castBlocks = NEL / 256;                 // 32768
    const int tcBlocks   = (DMODEL * KDIM) / 256;     // 4096
    const int gemmBlocks = (MTOT / 32) * (DMODEL / 64) / 8;  // 512
    const int attnBlocks = (BATCH * NH * (SEQ / 16)) / 8;    // 1024

    // Q projection (output pre-scaled by 1/sqrt(dk)*log2(e) for exp2 softmax)
    cast_bf16_kernel<<<castBlocks, 256, 0, stream>>>(query, Xb, NEL);
    transpose_cast_kernel<<<tcBlocks, 256, 0, stream>>>(Wq, Wt);
    gemm_wmma_kernel<0><<<gemmBlocks, 256, 0, stream>>>(Xb, Wt, bq, (void*)Qp, ALPHA_Q);

    // K projection
    cast_bf16_kernel<<<castBlocks, 256, 0, stream>>>(key, Xb, NEL);
    transpose_cast_kernel<<<tcBlocks, 256, 0, stream>>>(Wk, Wt);
    gemm_wmma_kernel<0><<<gemmBlocks, 256, 0, stream>>>(Xb, Wt, bk, (void*)Kp, 1.0f);

    // V projection (stored transposed per head)
    cast_bf16_kernel<<<castBlocks, 256, 0, stream>>>(value, Xb, NEL);
    transpose_cast_kernel<<<tcBlocks, 256, 0, stream>>>(Wv, Wt);
    gemm_wmma_kernel<1><<<gemmBlocks, 256, 0, stream>>>(Xb, Wt, bv, (void*)Vt, 1.0f);

    // Attention
    flash_attn_kernel<<<attnBlocks, 256, 0, stream>>>(Qp, Kp, Vt, Cx);

    // Output projection -> f32 d_out
    transpose_cast_kernel<<<tcBlocks, 256, 0, stream>>>(Wo, Wt);
    gemm_wmma_kernel<2><<<gemmBlocks, 256, 0, stream>>>(Cx, Wt, bo, d_out, 1.0f);
}
